// CausalSelfAttention_24043226923710
// MI455X (gfx1250) — matrix operand order repaired and checked by reading, not compiled, not run on hardware
//
#include <hip/hip_runtime.h>

// B=4, T=2048, D=1024, H=16, HD=64
typedef __attribute__((ext_vector_type(16))) __bf16        v16bf;
typedef __attribute__((ext_vector_type(8)))  float         v8f;
typedef __attribute__((ext_vector_type(8)))  unsigned int  v8u;
typedef __attribute__((ext_vector_type(4)))  __bf16        bf4;
typedef __attribute__((ext_vector_type(4)))  unsigned int  v4u;
typedef __attribute__((ext_vector_type(8)))  int           v8i;
typedef __attribute__((ext_vector_type(4)))  int           v4i;

#define WMMA_BF16(A, B, C) \
  __builtin_amdgcn_wmma_f32_16x16x32_bf16(false, (A), false, (B), (short)0, (C), false, false)

#if __has_builtin(__builtin_amdgcn_tensor_load_to_lds) && \
    __has_builtin(__builtin_amdgcn_s_wait_tensorcnt)
#define HAVE_TDM 1
#else
#define HAVE_TDM 0
#endif

#if HAVE_TDM
// Tensor Data Mover: DMA a 32(key) x 64(hd) bf16 tile (row stride 3072 elems)
// from global into LDS. 2D D# per ISA ch.8: group0 = {count|flags, lds_addr,
// global_addr lo, global_addr hi | type=2}; group1 packs data_size=2B,
// tensor_dim0=64, tensor_dim1=32, tile_dim0=64, tile_dim1=32, dim0_stride=3072.
// workgroup_mask = 0 (not in a cluster). Tracked on TENSORcnt.
// This toolchain exposes the 6-arg builtin:
//   (uint32x4 g0, int32x8 g1, int32x4 g2, int32x4 g3, int32x8 g4, i32 cpol)
__device__ __forceinline__ void tdm_load_k32x64(const __bf16* gsrc,
                                                unsigned int lds_byte) {
  unsigned long long ga = (unsigned long long)(uintptr_t)gsrc;
  v4u g0 = { 1u,                                    // count=1, no gather
             lds_byte,                              // LDS byte address
             (unsigned int)ga,                      // global_addr[31:0]
             (unsigned int)((ga >> 32) & 0x01ffffffu) | 0x80000000u }; // type=2
  v8i g1 = { 0x00010000,                            // data_size = 2 bytes
             (int)(64u << 16),                      // tensor_dim0 = 64
             (int)(32u << 16),                      // tensor_dim1 = 32
             (int)(64u << 16),                      // tile_dim0   = 64
             32,                                    // tile_dim1   = 32
             3072,                                  // tensor_dim0_stride
             0, 0 };
  v4i gz4 = { 0, 0, 0, 0 };                         // groups 2/3 unused (2D)
  v8i gz8 = { 0, 0, 0, 0, 0, 0, 0, 0 };
  __builtin_amdgcn_tensor_load_to_lds(g0, g1, gz4, gz4, gz8, 0);
}
#endif

// ---------------------------------------------------------------------------
// fp32 -> bf16 conversion, 4 elems/thread, 128-bit loads
// ---------------------------------------------------------------------------
__global__ __launch_bounds__(256) void f32_to_bf16_kernel(
    const float* __restrict__ in, __bf16* __restrict__ out, int n) {
  int i = (blockIdx.x * 256 + threadIdx.x) * 4;
  if (i + 3 < n) {
    float4 f = *(const float4*)(in + i);
    bf4 o = { (__bf16)f.x, (__bf16)f.y, (__bf16)f.z, (__bf16)f.w };
    *(bf4*)(out + i) = o;
  }
}

// ---------------------------------------------------------------------------
// GEMM: C[M,N] = A[M,K](bf16) * B[K,N](bf16) + bias, optional pad-row scale.
// Block tile 128x128, Kstep 32. 256 thr = 8 waves in a 4(M) x 2(N) grid;
// each wave owns 32x64 = 2x4 WMMA accumulators (f32).
// ---------------------------------------------------------------------------
template <bool OUT_BF16>
__global__ __launch_bounds__(256) void gemm_bf16_wmma(
    const __bf16* __restrict__ A, const __bf16* __restrict__ Bw,
    const float* __restrict__ bias, const int* __restrict__ pad,
    void* __restrict__ Cout, int M, int N, int K) {
  __shared__ unsigned int As32[128 * 16];   // [row][k-pair]
  __shared__ __bf16      Bs[128 * 32];      // [n][k]
  const unsigned int* Bs32 = (const unsigned int*)Bs;

  const int tid  = threadIdx.x;
  const int lane = tid & 31;
  const int wv   = tid >> 5;
  const int wm   = wv & 3;         // wave row  (4)
  const int wn   = wv >> 2;        // wave col  (2)
  const int hf   = lane >> 4;      // lane half (0/1)
  const int nl   = lane & 15;
  const int m0   = blockIdx.y * 128;
  const int n0   = blockIdx.x * 128;

  v8f zero8 = {};
  v8f acc[2][4];
#pragma unroll
  for (int mt = 0; mt < 2; ++mt)
#pragma unroll
    for (int nt = 0; nt < 4; ++nt) acc[mt][nt] = zero8;

  for (int k0 = 0; k0 < K; k0 += 32) {
    __syncthreads();
    // --- A tile: 128 rows x 32 k (8 KB) as 512 x 16B chunks, 2 per thread
#pragma unroll
    for (int c = 0; c < 2; ++c) {
      int chunk = tid + 256 * c;
      int row = chunk >> 2, seg = chunk & 3;
      uint4 d = *(const uint4*)(A + (size_t)(m0 + row) * K + k0 + seg * 8);
      *(uint4*)(&As32[row * 16 + seg * 4]) = d;
      if (c == 0 && k0 + 32 < K)  // speculative prefetch of next A k-tile
        __builtin_prefetch(A + (size_t)(m0 + row) * K + k0 + 32 + seg * 8, 0, 0);
    }
    // --- B tile: 32 k x 128 n, stored transposed [n][k]
    {
      int kr = tid >> 3;        // 0..31
      int ns = tid & 7;         // 0..7  -> 16 n each
      const __bf16* src = Bw + (size_t)(k0 + kr) * N + n0 + ns * 16;
      uint4 d0 = *(const uint4*)src;
      uint4 d1 = *(const uint4*)(src + 8);
      if (k0 + 32 < K)
        __builtin_prefetch(src + (size_t)32 * N, 0, 0);
      const __bf16* e0 = (const __bf16*)&d0;
      const __bf16* e1 = (const __bf16*)&d1;
#pragma unroll
      for (int i = 0; i < 8; ++i) {
        Bs[(ns * 16 + i) * 32 + kr]     = e0[i];
        Bs[(ns * 16 + 8 + i) * 32 + kr] = e1[i];
      }
    }
    __syncthreads();

    // --- fragments (A: 16x32, m=lane%16, K=(p/4)*16+half*8+2*(p&3))
    v16bf afr[2], bfr[4];
#pragma unroll
    for (int mt = 0; mt < 2; ++mt) {
      v8u raw;
      int m = wm * 32 + mt * 16 + nl;
#pragma unroll
      for (int p = 0; p < 8; ++p) {
        int kk = (p >> 2) * 16 + hf * 8 + (p & 3) * 2;
        raw[p] = As32[m * 16 + (kk >> 1)];
      }
      afr[mt] = __builtin_bit_cast(v16bf, raw);
    }
#pragma unroll
    for (int nt = 0; nt < 4; ++nt) {   // B: 32x16, n=lane%16, K=(p/4)*16+half*8+2*(p&3)
      v8u raw;
      int n = wn * 64 + nt * 16 + nl;
#pragma unroll
      for (int p = 0; p < 8; ++p) {
        int kk = (p >> 2) * 16 + hf * 8 + (p & 3) * 2;
        raw[p] = Bs32[n * 16 + (kk >> 1)];
      }
      bfr[nt] = __builtin_bit_cast(v16bf, raw);
    }
#pragma unroll
    for (int mt = 0; mt < 2; ++mt)
#pragma unroll
      for (int nt = 0; nt < 4; ++nt)
        acc[mt][nt] = WMMA_BF16(afr[mt], bfr[nt], acc[mt][nt]);
  }

  // --- epilogue: C layout row = r + 8*half, col = lane%16
#pragma unroll
  for (int mt = 0; mt < 2; ++mt)
#pragma unroll
    for (int nt = 0; nt < 4; ++nt)
#pragma unroll
      for (int r = 0; r < 8; ++r) {
        int row = m0 + wm * 32 + mt * 16 + r + 8 * hf;
        int col = n0 + wn * 64 + nt * 16 + nl;
        float v = acc[mt][nt][r] + bias[col];
        if (pad) v *= (float)pad[row];
        if (OUT_BF16)
          ((__bf16*)Cout)[(size_t)row * N + col] = (__bf16)v;
        else
          ((float*)Cout)[(size_t)row * N + col] = v;
      }
}

// ---------------------------------------------------------------------------
// Flash attention over packed qkv [B*T, 3072] bf16 (q|k|v each 1024 wide).
// BLOCK_Q=128 (8 waves x 16 rows), BK=32 keys/iter, HD=64, scale=1/8.
// K tiles: double-buffered in LDS, streamed by the Tensor Data Mover
// (wave 0 issues tensor_load_to_lds for tile i+1 while everyone computes on
// tile i; completion via s_wait_tensorcnt, in-order per wave). V tiles go
// through VMEM with a transposed LDS store (TDM cannot transpose).
// ---------------------------------------------------------------------------
__global__ __launch_bounds__(256) void attn_flash_wmma(
    const __bf16* __restrict__ qkv, const int* __restrict__ amask,
    __bf16* __restrict__ outb, int T) {
  __shared__ __bf16 Ks[2 * 32 * 64]; // double-buffered K [key][hd]
  __shared__ __bf16 Vs[64 * 32];     // [hd][key]  (transposed)
  __shared__ __bf16 Ps[8 * 16 * 32]; // per-wave P tile [16 q][32 key]
  const unsigned int* Vs32 = (const unsigned int*)Vs;
  const unsigned int* Ps32 = (const unsigned int*)Ps;

  const int tid  = threadIdx.x;
  const int lane = tid & 31;
  const int wv   = tid >> 5;
  const int hf   = lane >> 4;
  const int nl   = lane & 15;
  const int q0   = blockIdx.x * 128;
  const int h    = blockIdx.y;
  const int b    = blockIdx.z;
  const size_t bt = (size_t)b * T;

  // Q fragments for this wave's 16 rows (hd 0..63 -> two 16x32 A-frags)
  v16bf qfr[2];
  {
    int t = q0 + wv * 16 + nl;
    const unsigned int* qrow =
        (const unsigned int*)(qkv + (bt + t) * 3072 + h * 64);
#pragma unroll
    for (int f = 0; f < 2; ++f) {
      v8u raw;
#pragma unroll
      for (int p = 0; p < 8; ++p) {
        int kk = f * 32 + (p >> 2) * 16 + hf * 8 + (p & 3) * 2;
        raw[p] = qrow[kk >> 1];
      }
      qfr[f] = __builtin_bit_cast(v16bf, raw);
    }
  }

  v8f zero8 = {};
  v8f accO[4];
#pragma unroll
  for (int nt = 0; nt < 4; ++nt) accO[nt] = zero8;
  float rm[8], rl[8];
#pragma unroll
  for (int r = 0; r < 8; ++r) { rm[r] = -3.0e38f; rl[r] = 0.0f; }

  const int kend = min(T, q0 + 128);

#if HAVE_TDM
  if (wv == 0)  // kick off DMA of the first K tile
    tdm_load_k32x64(qkv + bt * 3072 + 1024 + h * 64,
                    (unsigned int)(unsigned long long)(uintptr_t)&Ks[0]);
#endif

  for (int k0 = 0; k0 < kend; k0 += 32) {
    const int buf = (k0 >> 5) & 1;
    __bf16* Kt = Ks + buf * 2048;
    __syncthreads();
    {
      int key = tid >> 3, seg = tid & 7;   // 32 keys x 8 segs of 8 bf16
      const __bf16* kbase = qkv + (bt + k0 + key) * 3072 + h * 64;
#if !HAVE_TDM
      uint4 kd = *(const uint4*)(kbase + 1024 + seg * 8);
      *(uint4*)(Kt + key * 64 + seg * 8) = kd;
#endif
      uint4 vd = *(const uint4*)(kbase + 2048 + seg * 8);
      const __bf16* ve = (const __bf16*)&vd;
#pragma unroll
      for (int i = 0; i < 8; ++i) Vs[(seg * 8 + i) * 32 + key] = ve[i];
      if (k0 + 32 < kend)  // speculative prefetch of next V tile
        __builtin_prefetch(qkv + (bt + k0 + 32 + key) * 3072 + 2048 + h * 64 +
                           seg * 8, 0, 0);
    }
#if HAVE_TDM
    if (wv == 0) {
      if (k0 + 32 < kend) {  // DMA next K tile into the other buffer
        tdm_load_k32x64(
            qkv + (bt + k0 + 32) * 3072 + 1024 + h * 64,
            (unsigned int)(unsigned long long)(uintptr_t)&Ks[(buf ^ 1) * 2048]);
        __builtin_amdgcn_s_wait_tensorcnt(1);  // in-order: current tile landed
      } else {
        __builtin_amdgcn_s_wait_tensorcnt(0);
      }
    }
#endif
    __syncthreads();
    const unsigned int* Kt32 = (const unsigned int*)Kt;

    // S = (Q K^T) * 0.125 : 2 key-tiles x 2 k-steps
    v8f s0 = zero8, s1 = zero8;
#pragma unroll
    for (int f = 0; f < 2; ++f) {
      v8u raw;
#pragma unroll
      for (int p = 0; p < 8; ++p) {
        int kk = f * 32 + (p >> 2) * 16 + hf * 8 + (p & 3) * 2;
        raw[p] = Kt32[nl * 32 + (kk >> 1)];           // key tile 0
      }
      s0 = WMMA_BF16(qfr[f], __builtin_bit_cast(v16bf, raw), s0);
#pragma unroll
      for (int p = 0; p < 8; ++p) {
        int kk = f * 32 + (p >> 2) * 16 + hf * 8 + (p & 3) * 2;
        raw[p] = Kt32[(16 + nl) * 32 + (kk >> 1)];    // key tile 1
      }
      s1 = WMMA_BF16(qfr[f], __builtin_bit_cast(v16bf, raw), s1);
    }

    // online softmax (rows r+8*half; cols across 16-lane halves)
    int col0 = k0 + nl, col1 = col0 + 16;
    int p0ok = amask[bt + col0], p1ok = amask[bt + col1];
#pragma unroll
    for (int r = 0; r < 8; ++r) {
      int rowq = q0 + wv * 16 + r + 8 * hf;
      float v0 = s0[r] * 0.125f, v1 = s1[r] * 0.125f;
      if (col0 > rowq || !p0ok) v0 = -3.0e38f;
      if (col1 > rowq || !p1ok) v1 = -3.0e38f;
      float mx = fmaxf(v0, v1);
#pragma unroll
      for (int off = 1; off < 16; off <<= 1)
        mx = fmaxf(mx, __shfl_xor(mx, off, 32));
      float nm = fmaxf(rm[r], mx);
      float e0 = __expf(v0 - nm), e1 = __expf(v1 - nm);
      float c  = __expf(rm[r] - nm);
      rm[r] = nm;
      float sum = e0 + e1;
#pragma unroll
      for (int off = 1; off < 16; off <<= 1) sum += __shfl_xor(sum, off, 32);
      rl[r] = rl[r] * c + sum;
#pragma unroll
      for (int nt = 0; nt < 4; ++nt) accO[nt][r] *= c;
      __bf16* prow = Ps + wv * 512 + (r + 8 * hf) * 32;
      prow[nl]      = (__bf16)e0;
      prow[16 + nl] = (__bf16)e1;
    }
    // same-wave DS RAW: wait for P stores before fragment gather
    asm volatile("s_wait_dscnt 0" ::: "memory");

    // O += P @ V  (A = P 16x32, B = V^T tiles over hd)
    v16bf pfr;
    {
      v8u raw;
#pragma unroll
      for (int p = 0; p < 8; ++p) {
        int kk = (p >> 2) * 16 + hf * 8 + (p & 3) * 2;
        raw[p] = Ps32[wv * 256 + nl * 16 + (kk >> 1)];
      }
      pfr = __builtin_bit_cast(v16bf, raw);
    }
#pragma unroll
    for (int nt = 0; nt < 4; ++nt) {
      v8u raw;
      int d = nt * 16 + nl;
#pragma unroll
      for (int p = 0; p < 8; ++p) {
        int kk = (p >> 2) * 16 + hf * 8 + (p & 3) * 2;
        raw[p] = Vs32[d * 16 + (kk >> 1)];
      }
      accO[nt] = WMMA_BF16(pfr, __builtin_bit_cast(v16bf, raw), accO[nt]);
    }
  }

  // normalize + write bf16 [B*T, 1024]
#pragma unroll
  for (int r = 0; r < 8; ++r) {
    float inv = 1.0f / rl[r];
    int t = q0 + wv * 16 + r + 8 * hf;
    __bf16* dst = outb + (bt + t) * 1024 + h * 64;
#pragma unroll
    for (int nt = 0; nt < 4; ++nt)
      dst[nt * 16 + nl] = (__bf16)(accO[nt][r] * inv);
  }
}

// ---------------------------------------------------------------------------
extern "C" void kernel_launch(void* const* d_in, const int* in_sizes, int n_in,
                              void* d_out, int out_size, void* d_ws,
                              size_t ws_size, hipStream_t stream) {
  const float* x     = (const float*)d_in[0];
  const int*   amask = (const int*)d_in[1];
  const float* Wqkv  = (const float*)d_in[2];
  const float* bqkv  = (const float*)d_in[3];
  const float* Wout  = (const float*)d_in[4];
  const float* bout  = (const float*)d_in[5];
  float* out = (float*)d_out;

  constexpr int Bb = 4, T = 2048, D = 1024, H = 16;
  const int M = Bb * T;  // 8192

  // workspace carve-up (bf16 staging, ~92 MB total)
  char* w = (char*)d_ws;
  __bf16* xbf    = (__bf16*)(w);
  __bf16* wqkvbf = (__bf16*)(w + 16777216);
  __bf16* woutbf = (__bf16*)(w + 23068672);
  __bf16* qkvbf  = (__bf16*)(w + 25165824);
  __bf16* attnbf = (__bf16*)(w + 75497472);

  f32_to_bf16_kernel<<<(M * D) / 1024, 256, 0, stream>>>(x, xbf, M * D);
  f32_to_bf16_kernel<<<(D * 3 * D) / 1024, 256, 0, stream>>>(Wqkv, wqkvbf,
                                                             D * 3 * D);
  f32_to_bf16_kernel<<<(D * D) / 1024, 256, 0, stream>>>(Wout, woutbf, D * D);

  // qkv = x @ Wqkv + bqkv        -> bf16 [M, 3072]
  gemm_bf16_wmma<true><<<dim3(3 * D / 128, M / 128), 256, 0, stream>>>(
      xbf, wqkvbf, bqkv, nullptr, qkvbf, M, 3 * D, D);

  // flash attention             -> bf16 [M, 1024]
  attn_flash_wmma<<<dim3(T / 128, H, Bb), 256, 0, stream>>>(qkvbf, amask,
                                                            attnbf, T);

  // out = attn @ Wout + bout, * pad  -> f32 d_out
  gemm_bf16_wmma<false><<<dim3(D / 128, M / 128), 256, 0, stream>>>(
      attnbf, woutbf, bout, amask, out, M, D, D);
}
